// LSTM_7782480741002
// MI455X (gfx1250) — matrix operand order repaired and checked by reading, not compiled, not run on hardware
//
#include <hip/hip_runtime.h>
#include <hip/hip_bf16.h>

typedef __attribute__((ext_vector_type(16))) __bf16 v16bf;
typedef __attribute__((ext_vector_type(8)))  __bf16 v8bf;
typedef __attribute__((ext_vector_type(8)))  float  v8f;

#define T_STEPS 512
#define H_DIM   256
#define NBLK    16

__device__ __forceinline__ float sigm(float v) { return 1.0f / (1.0f + __expf(-v)); }

// ---------------------------------------------------------------------------
// Init: bf16-convert recurrent weights, build P[g][digit][h] = Wx_g @ emb^T,
// zero h0/c and the grid-barrier counter (re-run every launch -> graph safe).
// ---------------------------------------------------------------------------
__global__ __launch_bounds__(256) void lstm_init(
    const float* __restrict__ emb,
    const float* __restrict__ Wxg, const float* __restrict__ Wxi,
    const float* __restrict__ Wxf, const float* __restrict__ Wxo,
    const float* __restrict__ Whg, const float* __restrict__ Whi,
    const float* __restrict__ Whf, const float* __restrict__ Who,
    __bf16* __restrict__ Whb, float* __restrict__ P,
    __bf16* __restrict__ h0, float* __restrict__ cT, unsigned* __restrict__ bar)
{
    const int gid = blockIdx.x * blockDim.x + threadIdx.x;
    const int stride = gridDim.x * blockDim.x;
    const float* Wh[4] = {Whg, Whi, Whf, Who};
    const float* Wx[4] = {Wxg, Wxi, Wxf, Wxo};

    for (int i = gid; i < H_DIM * H_DIM; i += stride) {
        #pragma unroll
        for (int g = 0; g < 4; ++g)
            Whb[(size_t)g * H_DIM * H_DIM + i] = (__bf16)Wh[g][i];
        h0[i] = (__bf16)0.0f;
        cT[i] = 0.0f;
    }
    for (int idx = gid; idx < 4 * 10 * H_DIM; idx += stride) {
        const int g   = idx / (10 * H_DIM);
        const int rem = idx - g * 10 * H_DIM;
        const int d   = rem / H_DIM;
        const int i   = rem - d * H_DIM;
        float s = 0.0f;
        #pragma unroll
        for (int e = 0; e < 6; ++e) s += Wx[g][i * 6 + e] * emb[d * 6 + e];
        P[idx] = s;
    }
    if (gid == 0) *bar = 0u;
}

// ---------------------------------------------------------------------------
// Main persistent kernel: 16 blocks x 256 threads (8 wave32 per block).
// Block m owns output rows [m*16, m*16+16); wave j owns column tiles {2j,2j+1}
// for ALL four gates. Wh fragments live in LDS (pre-swizzled A layout).
// h ping-pongs between two bf16 global buffers; grid barrier per timestep.
// ---------------------------------------------------------------------------
__global__ __launch_bounds__(256, 1) void lstm_main(
    const int* __restrict__ x,
    const float* __restrict__ bg, const float* __restrict__ bi,
    const float* __restrict__ bfb, const float* __restrict__ bo,
    const __bf16* __restrict__ Whb, const float* __restrict__ P,
    __bf16* __restrict__ h0buf, __bf16* __restrict__ h1buf,
    float* __restrict__ cT, float* __restrict__ hF,
    const float* __restrict__ Wp, const float* __restrict__ bp,
    float* __restrict__ out, unsigned* __restrict__ bar)
{
    __shared__ __attribute__((aligned(64))) __bf16 ldsA[4 * 8 * 32 * 16]; // 32 KB

    const int m    = blockIdx.x;        // row tile 0..15
    const int wave = threadIdx.x >> 5;
    const int lane = threadIdx.x & 31;
    const int lmod = lane & 15;
    const int lhi  = lane >> 4;

    // ---- preload A fragments (4 gates x 8 k-slabs) into LDS, swizzled so
    // each lane later does one contiguous 32B LDS read per fragment.
    for (int s = wave; s < 32; s += 8) {          // s = g*8 + kk
        const int g  = s >> 3, kk = s & 7;
        const int row = m * 16 + lmod;
        const int k0  = kk * 32 + lhi * 16;       // A layout: 16 contiguous K
        const __bf16* src = Whb + ((size_t)(g * H_DIM + row)) * H_DIM + k0;
        v8bf c0 = *(const v8bf*)(src);
        v8bf c1 = *(const v8bf*)(src + 8);
        __bf16* dst = &ldsA[((s * 32) + lane) * 16];
        *(v8bf*)dst       = c0;
        *(v8bf*)(dst + 8) = c1;
    }
    __syncthreads();

    const __bf16* hin  = h0buf;
    __bf16*       hout = h1buf;
    const int colA = (wave * 2 + 0) * 16 + lmod;  // batch column for n-tile A
    const int colB = (wave * 2 + 1) * 16 + lmod;  // batch column for n-tile B
    const int mrow = m * 16 + lhi * 8;            // first of this lane's 8 rows

    for (int t = 0; t < T_STEPS; ++t) {
        v8f acc[2][4] = {};                       // [ntile][gate] f32 accum

        const __bf16* hA = hin + (size_t)colA * H_DIM;
        const __bf16* hB = hin + (size_t)colB * H_DIM;
        const int kb = lhi * 16;                  // B layout: 16 contiguous K

        // software pipeline: B tiles for slab kk prefetched during slab kk-1
        v16bf bA = *(const v16bf*)(hA + kb);
        v16bf bB = *(const v16bf*)(hB + kb);

        #pragma unroll
        for (int kk = 0; kk < 8; ++kk) {          // K = 256 in 8 x 32 slabs
            // load ALL four gate A-fragments first so the dscnt waits
            // coalesce and the 8 WMMAs below issue back-to-back
            const __bf16* lp = &ldsA[(size_t)(kk * 32 + lane) * 16];
            v16bf a0 = *(const v16bf*)(lp + 0 * 8 * 32 * 16);
            v16bf a1 = *(const v16bf*)(lp + 1 * 8 * 32 * 16);
            v16bf a2 = *(const v16bf*)(lp + 2 * 8 * 32 * 16);
            v16bf a3 = *(const v16bf*)(lp + 3 * 8 * 32 * 16);

            v16bf bAc = bA, bBc = bB;
            if (kk < 7) {                         // prefetch next slab's B tiles
                const int kn = (kk + 1) * 32 + kb;
                bA = *(const v16bf*)(hA + kn);
                bB = *(const v16bf*)(hB + kn);
            }

            acc[0][0] = __builtin_amdgcn_wmma_f32_16x16x32_bf16(
                false, a0, false, bAc, (short)0, acc[0][0], false, false);
            acc[1][0] = __builtin_amdgcn_wmma_f32_16x16x32_bf16(
                false, a0, false, bBc, (short)0, acc[1][0], false, false);
            acc[0][1] = __builtin_amdgcn_wmma_f32_16x16x32_bf16(
                false, a1, false, bAc, (short)0, acc[0][1], false, false);
            acc[1][1] = __builtin_amdgcn_wmma_f32_16x16x32_bf16(
                false, a1, false, bBc, (short)0, acc[1][1], false, false);
            acc[0][2] = __builtin_amdgcn_wmma_f32_16x16x32_bf16(
                false, a2, false, bAc, (short)0, acc[0][2], false, false);
            acc[1][2] = __builtin_amdgcn_wmma_f32_16x16x32_bf16(
                false, a2, false, bBc, (short)0, acc[1][2], false, false);
            acc[0][3] = __builtin_amdgcn_wmma_f32_16x16x32_bf16(
                false, a3, false, bAc, (short)0, acc[0][3], false, false);
            acc[1][3] = __builtin_amdgcn_wmma_f32_16x16x32_bf16(
                false, a3, false, bBc, (short)0, acc[1][3], false, false);
        }

        #pragma unroll
        for (int nn = 0; nn < 2; ++nn) {
            const int b = nn ? colB : colA;
            const int d = x[(size_t)b * T_STEPS + t];     // digit 0..9
            // faithful to reference broadcast: bias indexed by batch column
            const float bgs = bg[b], bis = bi[b], bfs = bfb[b], bos = bo[b];
            const float* Pg = P + (0 * 10 + d) * H_DIM + mrow;
            const float* Pi = P + (1 * 10 + d) * H_DIM + mrow;
            const float* Pf = P + (2 * 10 + d) * H_DIM + mrow;
            const float* Po = P + (3 * 10 + d) * H_DIM + mrow;
            float* cptr = cT + (size_t)b * H_DIM + mrow;
            v8bf  hv;
            float hw[8];
            #pragma unroll
            for (int r = 0; r < 8; ++r) {
                const float pg = acc[nn][0][r] + Pg[r] + bgs;
                const float pi = acc[nn][1][r] + Pi[r] + bis;
                const float pf = acc[nn][2][r] + Pf[r] + bfs;
                const float po = acc[nn][3][r] + Po[r] + bos;
                const float gg = tanhf(pg);
                const float ii = sigm(pi);
                const float ff = sigm(pf);
                const float oo = sigm(po);
                const float cn = gg * ii + cptr[r] * ff;
                cptr[r] = cn;
                const float hh = tanhf(cn) * oo;
                hv[r] = (__bf16)hh;
                hw[r] = hh;
            }
            *(v8bf*)(hout + (size_t)b * H_DIM + mrow) = hv;   // contiguous 16B
            if (t == T_STEPS - 1) {
                float* hp = hF + (size_t)b * H_DIM + mrow;
                #pragma unroll
                for (int r = 0; r < 8; ++r) hp[r] = hw[r];
            }
        }

        // ---- grid-wide barrier (monotonic counter; init kernel zeroed it)
        __threadfence();
        __syncthreads();
        if (threadIdx.x == 0) {
            __hip_atomic_fetch_add(bar, 1u, __ATOMIC_ACQ_REL, __HIP_MEMORY_SCOPE_AGENT);
            const unsigned want = (unsigned)(NBLK * (t + 1));
            while (__hip_atomic_load(bar, __ATOMIC_ACQUIRE, __HIP_MEMORY_SCOPE_AGENT) < want)
                __builtin_amdgcn_s_sleep(1);
        }
        __syncthreads();

        const __bf16* tmp = hin; hin = hout; hout = (__bf16*)tmp;
    }

    // ---- epilogue: out[b][c] = Wp[c,:] . hF[b,:] + bp[c]  (tiny: 0.65 MFLOP)
    const int tid = threadIdx.x;
    if (tid < 160) {
        const int batch = m * 16 + tid / 10;
        const int cls   = tid - (tid / 10) * 10;
        const float* hp = hF + (size_t)batch * H_DIM;
        const float* wp = Wp + (size_t)cls * H_DIM;
        float s = bp[cls];
        for (int k = 0; k < H_DIM; ++k) s += wp[k] * hp[k];
        out[batch * 10 + cls] = s;
    }
}

// ---------------------------------------------------------------------------
extern "C" void kernel_launch(void* const* d_in, const int* in_sizes, int n_in,
                              void* d_out, int out_size, void* d_ws, size_t ws_size,
                              hipStream_t stream) {
    const int*   x   = (const int*)  d_in[0];
    const float* emb = (const float*)d_in[1];
    const float* Wxg = (const float*)d_in[2];
    const float* Whg = (const float*)d_in[3];
    const float* bg  = (const float*)d_in[4];
    const float* Wxi = (const float*)d_in[5];
    const float* Whi = (const float*)d_in[6];
    const float* bi  = (const float*)d_in[7];
    const float* Wxf = (const float*)d_in[8];
    const float* Whf = (const float*)d_in[9];
    const float* bfp = (const float*)d_in[10];
    const float* Wxo = (const float*)d_in[11];
    const float* Who = (const float*)d_in[12];
    const float* bo  = (const float*)d_in[13];
    const float* Wp  = (const float*)d_in[14];
    const float* bp  = (const float*)d_in[15];

    char* ws = (char*)d_ws;
    __bf16*  Whb = (__bf16*)ws;   ws += (size_t)4 * 256 * 256 * 2;  // 512 KB
    float*   P   = (float*)ws;    ws += (size_t)4 * 10 * 256 * 4;   //  40 KB
    __bf16*  h0  = (__bf16*)ws;   ws += (size_t)256 * 256 * 2;      // 128 KB
    __bf16*  h1  = (__bf16*)ws;   ws += (size_t)256 * 256 * 2;      // 128 KB
    float*   cT  = (float*)ws;    ws += (size_t)256 * 256 * 4;      // 256 KB
    float*   hF  = (float*)ws;    ws += (size_t)256 * 256 * 4;      // 256 KB
    unsigned* bar = (unsigned*)ws;

    lstm_init<<<256, 256, 0, stream>>>(emb, Wxg, Wxi, Wxf, Wxo,
                                       Whg, Whi, Whf, Who,
                                       Whb, P, h0, cT, bar);
    lstm_main<<<NBLK, 256, 0, stream>>>(x, bg, bi, bfp, bo, Whb, P,
                                        h0, h1, cT, hF, Wp, bp,
                                        (float*)d_out, bar);
}
